// CriticalBrainDynamics_48550310314464
// MI455X (gfx1250) — 8-bit matrix operand order repaired and checked by reading, compiled, not run on hardware
//
#include <hip/hip_runtime.h>
#include <stdint.h>

#define NN      4096
#define BB      1024
#define NSTEPS  10

#define FP8_ONE 0x38

typedef __attribute__((ext_vector_type(16))) int   v16i;
typedef __attribute__((ext_vector_type(8)))  float v8f;

__global__ void cbd_init_state(float* __restrict__ v, unsigned char* __restrict__ s0,
                               const float* __restrict__ mp, const float* __restrict__ ext) {
  int i = blockIdx.x * blockDim.x + threadIdx.x;
  int n = i & (NN - 1);
  v[i] = mp[n] + ext[i];
  s0[i] = 0;
}

__global__ void cbd_init_conn(unsigned char* __restrict__ Cu, const float* __restrict__ Cf) {
  size_t i = (size_t)blockIdx.x * blockDim.x + threadIdx.x;
  Cu[i] = (Cf[i] != 0.0f) ? (unsigned char)FP8_ONE : (unsigned char)0;
}

__global__ void cbd_init_refr(float* __restrict__ refr, const float* __restrict__ rin) {
  int i = blockIdx.x * blockDim.x + threadIdx.x;
  refr[i] = rin[i];
}

__device__ __forceinline__ v16i cbd_load_a(const unsigned char* aRow, int k0) {
  const uint4* pA = (const uint4*)(aRow + k0);
  v16i A;
#pragma unroll
  for (int g = 0; g < 4; ++g) {
    uint4 a = pA[g * 2];
    A[g * 4 + 0] = (int)a.x;
    A[g * 4 + 1] = (int)a.y;
    A[g * 4 + 2] = (int)a.z;
    A[g * 4 + 3] = (int)a.w;
  }
  return A;
}

__device__ __forceinline__ v16i cbd_load_b(const unsigned char* bRow, int k0) {
  const uint4* pB = (const uint4*)(bRow + k0);
  v16i B;
#pragma unroll
  for (int g = 0; g < 4; ++g) {
    uint4 b = pB[g * 2];
    B[g * 4 + 0] = (int)b.x;
    B[g * 4 + 1] = (int)b.y;
    B[g * 4 + 2] = (int)b.z;
    B[g * 4 + 3] = (int)b.w;
  }
  return B;
}

__global__ __launch_bounds__(256) void cbd_step(
    const unsigned char* __restrict__ sPrev, unsigned char* __restrict__ sNext,
    const unsigned char* __restrict__ Cu, float* __restrict__ v,
    const float* __restrict__ thr, const float* __restrict__ refr,
    int* __restrict__ anyFlag, float* __restrict__ out, int last) {
  const int lane = threadIdx.x & 31;
  const int wave = threadIdx.x >> 5;
  const int half = lane >> 4;
  const int l15  = lane & 15;
  const int mBase = blockIdx.y * 32;
  const int nWave = blockIdx.x * 512 + wave * 64;

  const unsigned char* aRow0 = sPrev + (size_t)(mBase + l15) * NN + half * 16;
  const unsigned char* aRow1 = aRow0 + (size_t)16 * NN;
  const unsigned char* bRow[4];
#pragma unroll
  for (int t = 0; t < 4; ++t)
    bRow[t] = Cu + (size_t)(nWave + t * 16 + l15) * NN + half * 16;

  v8f acc0[4], acc1[4];
#pragma unroll
  for (int t = 0; t < 4; ++t) {
    acc0[t] = (v8f){0.f, 0.f, 0.f, 0.f, 0.f, 0.f, 0.f, 0.f};
    acc1[t] = (v8f){0.f, 0.f, 0.f, 0.f, 0.f, 0.f, 0.f, 0.f};
  }

#pragma unroll 1
  for (int k0 = 0; k0 < NN; k0 += 128) {
    v16i A0 = cbd_load_a(aRow0, k0);
    v16i A1 = cbd_load_a(aRow1, k0);
    if (k0 + 128 < NN)
      __builtin_prefetch(aRow0 + k0 + 128, 0, 1);

#pragma unroll
    for (int t = 0; t < 4; ++t) {
      v16i Bf = cbd_load_b(bRow[t], k0);
      acc0[t] = __builtin_amdgcn_wmma_f32_16x16x128_fp8_fp8(
          A0, Bf, (short)0, acc0[t], false, false);
      acc1[t] = __builtin_amdgcn_wmma_f32_16x16x128_fp8_fp8(
          A1, Bf, (short)0, acc1[t], false, false);
    }
  }

#pragma unroll
  for (int t = 0; t < 4; ++t) {
    const int col = nWave + t * 16 + l15;
    const float th = thr[col];
    const float rf = refr[col];
    const bool refrOk = (rf == 0.0f);
    int spikedAny = 0;
#pragma unroll
    for (int h = 0; h < 2; ++h) {
      const v8f* acc = h ? acc1 : acc0;
#pragma unroll
      for (int j = 0; j < 8; ++j) {
        const int row = mBase + h * 16 + half * 8 + j;
        const size_t idx = (size_t)row * NN + col;
        float vv = v[idx] + 0.1f * acc[t][j];
        const bool m = (vv > th) && refrOk;
        sNext[idx] = m ? (unsigned char)FP8_ONE : (unsigned char)0;
        vv = m ? 0.0f : vv;
        vv *= 0.95f;
        v[idx] = vv;
        spikedAny |= (int)m;
        if (last) out[idx] = m ? 1.0f : 0.0f;
      }
    }
    if (spikedAny) atomicOr(&anyFlag[col], 1);
  }
}

__global__ void cbd_refr(float* __restrict__ refr, const int* __restrict__ anyFlag) {
  int n = blockIdx.x * blockDim.x + threadIdx.x;
  float r = refr[n];
  if (anyFlag[n] != 0) r = 3.0f;
  r -= 1.0f;
  r = fminf(fmaxf(r, 0.0f), 10.0f);
  refr[n] = r;
}

extern "C" void kernel_launch(void* const* d_in, const int* in_sizes, int n_in,
                              void* d_out, int out_size, void* d_ws, size_t ws_size,
                              hipStream_t stream) {
  (void)in_sizes; (void)n_in; (void)out_size; (void)ws_size;
  const float* ext  = (const float*)d_in[0];
  const float* conn = (const float*)d_in[1];
  const float* mp   = (const float*)d_in[2];
  const float* thr  = (const float*)d_in[3];
  const float* rin  = (const float*)d_in[4];
  float* out = (float*)d_out;

  char* ws = (char*)d_ws;
  unsigned char* Cu  = (unsigned char*)ws;  ws += (size_t)NN * NN;
  float*         v   = (float*)ws;          ws += (size_t)BB * NN * 4;
  unsigned char* sA  = (unsigned char*)ws;  ws += (size_t)BB * NN;
  unsigned char* sB  = (unsigned char*)ws;  ws += (size_t)BB * NN;
  float*         rf  = (float*)ws;          ws += (size_t)NN * 4;
  int*           anyF= (int*)ws;            ws += (size_t)NN * 4;

  cbd_init_state<<<(BB * NN) / 256, 256, 0, stream>>>(v, sA, mp, ext);
  cbd_init_conn<<<(int)(((size_t)NN * NN) / 256), 256, 0, stream>>>(Cu, conn);
  cbd_init_refr<<<NN / 256, 256, 0, stream>>>(rf, rin);

  unsigned char* sCur = sA;
  unsigned char* sNxt = sB;
  for (int t = 0; t < NSTEPS; ++t) {
    hipMemsetAsync(anyF, 0, NN * sizeof(int), stream);
    dim3 grid(NN / 512, BB / 32);
    cbd_step<<<grid, 256, 0, stream>>>(sCur, sNxt, Cu, v, thr, rf, anyF, out,
                                       t == NSTEPS - 1 ? 1 : 0);
    if (t < NSTEPS - 1)
      cbd_refr<<<NN / 256, 256, 0, stream>>>(rf, anyF);
    unsigned char* tmp = sCur; sCur = sNxt; sNxt = tmp;
  }
}
